// AMGCN_cls_GraphConv_146028888441
// MI455X (gfx1250) — matrix operand order repaired and checked by reading, not compiled, not run on hardware
//
#include <hip/hip_runtime.h>
#include <hip/hip_bf16.h>
#include <math.h>

#define NPTS 4096
#define KMAIN 20
#define KPLANE 50

typedef __attribute__((ext_vector_type(16))) __bf16 v16bf;
typedef __attribute__((ext_vector_type(8)))  float  v8f;

__device__ __forceinline__ unsigned short f2bf(float f) {
    unsigned u = __float_as_uint(f);
    unsigned r = (u + 0x7FFFu + ((u >> 16) & 1u)) >> 16;
    return (unsigned short)r;
}

// ---------------- small elementwise kernels ----------------

__global__ void fzero_kernel(float* p, int n) {
    int i = blockIdx.x * blockDim.x + threadIdx.x;
    if (i < n) p[i] = 0.0f;
}

__global__ void f32_to_bf16_kernel(const float* in, unsigned short* out, int n) {
    int i = blockIdx.x * blockDim.x + threadIdx.x;
    if (i < n) out[i] = f2bf(in[i]);
}

// w: [K,N] row-major f32  ->  wt: [N,K] row-major bf16
__global__ void transpose_to_bf16_kernel(const float* w, unsigned short* wt, int K, int N) {
    int idx = blockIdx.x * blockDim.x + threadIdx.x;
    if (idx < K * N) {
        int nIdx = idx / K, kIdx = idx % K;
        wt[nIdx * K + kIdx] = f2bf(w[kIdx * N + nIdx]);
    }
}

// ---------------- WMMA bf16 GEMM: C[M,N] = A[M,K] @ Wt[N,K]^T (+bias) ----------------
// One wave computes a 32x64 output tile (2 M-frags x 4 N-frags, 8 WMMAs per k-step
// sharing 6 fragment loads -> ~2.7x less L2 traffic per WMMA than 1 tile/wave).
__global__ __launch_bounds__(32) void wmma_gemm_bf16(
    const unsigned short* Aus, int lda,
    const unsigned short* Wus, int ldw,
    const float* bias,
    float* C, int ldc,
    int K)
{
    const __bf16* A  = reinterpret_cast<const __bf16*>(Aus);
    const __bf16* Wt = reinterpret_cast<const __bf16*>(Wus);
    int lane = threadIdx.x;
    int half = lane >> 4;
    int l    = lane & 15;
    int ntile = blockIdx.x * 64;
    int mtile = blockIdx.y * 32;

    const __bf16* Arow0 = A + (long)(mtile + l) * lda;        // M-frag 0: rows mtile..mtile+15
    const __bf16* Arow1 = A + (long)(mtile + 16 + l) * lda;   // M-frag 1: rows mtile+16..+31
    const __bf16* Wrow0 = Wt + (long)(ntile +  0 + l) * ldw;  // N-frags 0..3
    const __bf16* Wrow1 = Wt + (long)(ntile + 16 + l) * ldw;
    const __bf16* Wrow2 = Wt + (long)(ntile + 32 + l) * ldw;
    const __bf16* Wrow3 = Wt + (long)(ntile + 48 + l) * ldw;

    v8f acc[2][4];
    for (int mi = 0; mi < 2; ++mi)
        for (int ni = 0; ni < 4; ++ni)
            for (int r = 0; r < 8; ++r) acc[mi][ni][r] = 0.0f;

    for (int k0 = 0; k0 < K; k0 += 32) {
        // A fragment layout (16-bit A 16x32): lanes 0-15 hold K 0..15,
        // lanes 16-31 hold K 16..31, M = lane&15.
        int ka = k0 + half * 16;
        v16bf a0, a1;
        for (int e = 0; e < 16; ++e) {
            a0[e] = Arow0[ka + e];
            a1[e] = Arow1[ka + e];
        }
        // B fragment (32x16): lanes 0-15 = K 0..15, lanes 16-31 = K 16..31, N = lane&15.
        int kb = k0 + half * 16;
        v16bf b0, b1, b2, b3;
        for (int e = 0; e < 16; ++e) {
            b0[e] = Wrow0[kb + e];
            b1[e] = Wrow1[kb + e];
            b2[e] = Wrow2[kb + e];
            b3[e] = Wrow3[kb + e];
        }
        if (k0 + 32 < K) {
            __builtin_prefetch(Arow0 + k0 + 32, 0, 0);
            __builtin_prefetch(Arow1 + k0 + 32, 0, 0);
            __builtin_prefetch(Wrow0 + k0 + 32, 0, 0);
            __builtin_prefetch(Wrow1 + k0 + 32, 0, 0);
            __builtin_prefetch(Wrow2 + k0 + 32, 0, 0);
            __builtin_prefetch(Wrow3 + k0 + 32, 0, 0);
        }
        acc[0][0] = __builtin_amdgcn_wmma_f32_16x16x32_bf16(false, a0, false, b0, (short)0, acc[0][0], false, false);
        acc[0][1] = __builtin_amdgcn_wmma_f32_16x16x32_bf16(false, a0, false, b1, (short)0, acc[0][1], false, false);
        acc[0][2] = __builtin_amdgcn_wmma_f32_16x16x32_bf16(false, a0, false, b2, (short)0, acc[0][2], false, false);
        acc[0][3] = __builtin_amdgcn_wmma_f32_16x16x32_bf16(false, a0, false, b3, (short)0, acc[0][3], false, false);
        acc[1][0] = __builtin_amdgcn_wmma_f32_16x16x32_bf16(false, a1, false, b0, (short)0, acc[1][0], false, false);
        acc[1][1] = __builtin_amdgcn_wmma_f32_16x16x32_bf16(false, a1, false, b1, (short)0, acc[1][1], false, false);
        acc[1][2] = __builtin_amdgcn_wmma_f32_16x16x32_bf16(false, a1, false, b2, (short)0, acc[1][2], false, false);
        acc[1][3] = __builtin_amdgcn_wmma_f32_16x16x32_bf16(false, a1, false, b3, (short)0, acc[1][3], false, false);
    }

    // C layout: element r of v8f -> row = tileRow + r + half*8, col = tileCol + (lane&15)
    for (int ni = 0; ni < 4; ++ni) {
        int col = ntile + 16 * ni + l;
        float bv = bias ? bias[col] : 0.0f;
        for (int mi = 0; mi < 2; ++mi) {
            int rbase = mtile + 16 * mi + half * 8;
            for (int r = 0; r < 8; ++r) {
                C[(long)(rbase + r) * ldc + col] = acc[mi][ni][r] + bv;
            }
        }
    }
}

// ---------------- fused kNN (distance + top-k selection), one block per query ----------------
__global__ void knn_kernel(const float* pts, int ldp, int d, int k, int* out) {
    __shared__ float q[64];
    __shared__ float rd[256];
    __shared__ int   ri[256];
    __shared__ int   winner;
    int i = blockIdx.x, t = threadIdx.x;
    if (t < d) q[t] = pts[(long)i * ldp + t];
    __syncthreads();

    float dist[16];
    for (int c = 0; c < 16; ++c) {
        int j = t + c * 256;
        const float* pj = pts + (long)j * ldp;
        float s = 0.0f;
        for (int e = 0; e < d; ++e) { float df = pj[e] - q[e]; s += df * df; }
        dist[c] = s;
    }
    unsigned sel = 0;
    for (int r = 0; r < k; ++r) {
        float bd = 3.4e38f; int bi = 0x7fffffff;
        for (int c = 0; c < 16; ++c) {
            if (sel & (1u << c)) continue;
            int j = t + c * 256;
            float dd = dist[c];
            if (dd < bd || (dd == bd && j < bi)) { bd = dd; bi = j; }
        }
        rd[t] = bd; ri[t] = bi;
        __syncthreads();
        for (int s = 128; s > 0; s >>= 1) {
            if (t < s) {
                float od = rd[t + s]; int oi = ri[t + s];
                if (od < rd[t] || (od == rd[t] && oi < ri[t])) { rd[t] = od; ri[t] = oi; }
            }
            __syncthreads();
        }
        if (t == 0) { winner = ri[0]; out[(long)i * k + r] = ri[0]; }
        __syncthreads();
        int w = winner;
        if ((w & 255) == t) sel |= 1u << (w >> 8);
        __syncthreads();
    }
}

// ---------------- GraphConv pieces ----------------

__global__ void degcount_kernel(const int* nbr, int total, float* deg) {
    int e = blockIdx.x * blockDim.x + threadIdx.x;
    if (e < total) atomicAdd(&deg[nbr[e]], 1.0f);
}

__global__ void rdeg_kernel(const float* deg, float* rdeg, int n) {
    int i = blockIdx.x * blockDim.x + threadIdx.x;
    if (i < n) rdeg[i] = rsqrtf(fmaxf(deg[i], 1.0f));
}

// out[i,:] = ( sum_j x[nbr[i,j],:]*rdeg[nbr[i,j]] ) * rsk @ w + b
__global__ void gconv_kernel(const int* nbr, int k, const float* x, int ldx,
                             const float* rdeg, const float* w, const float* b,
                             int din, int dout, float* out, int ldo, float rsk)
{
    __shared__ float agg[64];
    int i = blockIdx.x, t = threadIdx.x;
    if (t < din) {
        float s = 0.0f;
        for (int j = 0; j < k; ++j) {
            int v = nbr[(long)i * k + j];
            s += x[(long)v * ldx + t] * rdeg[v];
        }
        agg[t] = s * rsk;
    }
    __syncthreads();
    if (t < dout) {
        float s = b[t];
        for (int e = 0; e < din; ++e) s += agg[e] * w[e * dout + t];
        out[(long)i * ldo + t] = s;
    }
}

// ---------------- misc dense kernels ----------------

__global__ void tnet_fc1_kernel(const float* pc, const float* w, const float* b, float* out) {
    int idx = blockIdx.x * blockDim.x + threadIdx.x;  // N*64
    if (idx < NPTS * 64) {
        int i = idx >> 6, c = idx & 63;
        float s = b[c];
        for (int e = 0; e < 3; ++e) s += pc[i * 3 + e] * w[e * 64 + c];
        out[idx] = fmaxf(s, 0.0f);
    }
}

__global__ void colmax_kernel(const float* A, int M, int N, float* out) {
    __shared__ float s[256];
    int c = blockIdx.x, t = threadIdx.x;
    float m = -3.4e38f;
    for (int r = t; r < M; r += 256) m = fmaxf(m, A[(long)r * N + c]);
    s[t] = m; __syncthreads();
    for (int st = 128; st > 0; st >>= 1) {
        if (t < st) s[t] = fmaxf(s[t], s[t + st]);
        __syncthreads();
    }
    if (t == 0) out[c] = s[0];
}

__global__ void tnet_head_qr_kernel(const float* maxv, const float* w3, const float* b3,
                                    const float* w4, const float* b4,
                                    float* qout, float* rout)
{
    __shared__ float t3[256];
    __shared__ float t9[9];
    int t = threadIdx.x;
    float s = b3[t];
    for (int e = 0; e < 1024; ++e) s += maxv[e] * w3[e * 256 + t];
    t3[t] = fmaxf(s, 0.0f);
    __syncthreads();
    if (t < 9) {
        float s2 = b4[t];
        for (int e = 0; e < 256; ++e) s2 += t3[e] * w4[e * 9 + t];
        t9[t] = s2;
    }
    __syncthreads();
    if (t == 0) {
        float A[3][3], Q[3][3];
        for (int i = 0; i < 3; ++i)
            for (int j = 0; j < 3; ++j) { A[i][j] = t9[i * 3 + j]; Q[i][j] = (i == j) ? 1.f : 0.f; }
        for (int kk = 0; kk < 2; ++kk) {
            float nrm = 0.f;
            for (int i = kk; i < 3; ++i) nrm += A[i][kk] * A[i][kk];
            nrm = sqrtf(nrm);
            if (nrm > 1e-20f) {
                float alpha = (A[kk][kk] >= 0.f) ? -nrm : nrm;
                float v[3] = {0.f, 0.f, 0.f};
                v[kk] = A[kk][kk] - alpha;
                for (int i = kk + 1; i < 3; ++i) v[i] = A[i][kk];
                float vn2 = 0.f;
                for (int i = 0; i < 3; ++i) vn2 += v[i] * v[i];
                if (vn2 > 1e-30f) {
                    for (int j = 0; j < 3; ++j) {
                        float dot = 0.f;
                        for (int i = kk; i < 3; ++i) dot += v[i] * A[i][j];
                        float f = 2.f * dot / vn2;
                        for (int i = kk; i < 3; ++i) A[i][j] -= f * v[i];
                    }
                    for (int i = 0; i < 3; ++i) {
                        float dot = 0.f;
                        for (int j = kk; j < 3; ++j) dot += Q[i][j] * v[j];
                        float f = 2.f * dot / vn2;
                        for (int j = kk; j < 3; ++j) Q[i][j] -= f * v[j];
                    }
                }
            }
        }
        for (int i = 0; i < 3; ++i)
            for (int j = 0; j < 3; ++j) {
                qout[i * 3 + j] = Q[i][j];
                rout[i * 3 + j] = (j >= i) ? A[i][j] : 0.f;
            }
    }
}

__global__ void rotate_project_kernel(const float* pc, const float* q, float* x,
                                      float* pxy, float* pyz, float* pxz)
{
    int i = blockIdx.x * blockDim.x + threadIdx.x;
    if (i < NPTS) {
        float v[3];
        for (int c = 0; c < 3; ++c)
            v[c] = pc[i*3+0]*q[0*3+c] + pc[i*3+1]*q[1*3+c] + pc[i*3+2]*q[2*3+c];
        x[i*3+0]=v[0]; x[i*3+1]=v[1]; x[i*3+2]=v[2];
        pxy[i*2+0]=v[0]; pxy[i*2+1]=v[1];
        pyz[i*2+0]=v[1]; pyz[i*2+1]=v[2];
        pxz[i*2+0]=v[0]; pxz[i*2+1]=v[2];
    }
}

__global__ void broadcast_cols_kernel(const float* vec, float* H, int ldh, int col0, int nc) {
    int idx = blockIdx.x * blockDim.x + threadIdx.x;
    if (idx < NPTS * nc) {
        int r = idx / nc, c = idx % nc;
        H[(long)r * ldh + col0 + c] = vec[c];
    }
}

// ---------------- GATv2 attention (one block per node) ----------------
__global__ void gatv2_kernel(const int* nbr, const float* fs, const float* fd,
                             const float* a, const float* bias, float* out)
{
    __shared__ float fdi[1024];
    __shared__ float red4[4][256];
    __shared__ float score[4][KMAIN];
    __shared__ float attn[4][KMAIN];
    int i = blockIdx.x, t = threadIdx.x;
    for (int q = 0; q < 4; ++q) fdi[t + 256 * q] = fd[(long)i * 1024 + t + 256 * q];
    __syncthreads();

    for (int j = 0; j < KMAIN; ++j) {
        int v = nbr[(long)i * KMAIN + j];
        float partial[4];
        for (int q = 0; q < 4; ++q) {
            int dcol = t + 256 * q;
            float e = fs[(long)v * 1024 + dcol] + fdi[dcol];
            e = (e > 0.f) ? e : 0.2f * e;
            partial[q] = e * a[q * 256 + t];
        }
        for (int q = 0; q < 4; ++q) red4[q][t] = partial[q];
        __syncthreads();
        for (int s = 128; s > 0; s >>= 1) {
            if (t < s) for (int q = 0; q < 4; ++q) red4[q][t] += red4[q][t + s];
            __syncthreads();
        }
        if (t < 4) score[t][j] = red4[t][0];
        __syncthreads();
    }
    if (t < 4) {
        float mx = -3.4e38f;
        for (int j = 0; j < KMAIN; ++j) mx = fmaxf(mx, score[t][j]);
        float sum = 0.f;
        for (int j = 0; j < KMAIN; ++j) { float e = expf(score[t][j] - mx); attn[t][j] = e; sum += e; }
        float inv = 1.f / sum;
        for (int j = 0; j < KMAIN; ++j) attn[t][j] *= inv;
    }
    __syncthreads();
    float acc[4] = {0.f, 0.f, 0.f, 0.f};
    for (int j = 0; j < KMAIN; ++j) {
        int v = nbr[(long)i * KMAIN + j];
        for (int q = 0; q < 4; ++q)
            acc[q] += attn[q][j] * fs[(long)v * 1024 + t + 256 * q];
    }
    for (int q = 0; q < 4; ++q)
        out[(long)i * 1024 + t + 256 * q] = acc[q] + bias[t + 256 * q];
}

// ---------------- classifier head ----------------
__global__ void head_kernel(const float* hg, const float* w1, const float* b1,
                            const float* w2, const float* b2,
                            const float* w3, const float* b3, float* out)
{
    __shared__ float h1[256];
    __shared__ float h2[64];
    int t = threadIdx.x;
    float s = b1[t];
    for (int e = 0; e < 1024; ++e) s += hg[e] * w1[e * 256 + t];
    h1[t] = fmaxf(s, 0.f);
    __syncthreads();
    if (t < 64) {
        float s2 = b2[t];
        for (int e = 0; e < 256; ++e) s2 += h1[e] * w2[e * 64 + t];
        h2[t] = fmaxf(s2, 0.f);
    }
    __syncthreads();
    if (t < 40) {
        float s3 = b3[t];
        for (int e = 0; e < 64; ++e) s3 += h2[e] * w3[e * 40 + t];
        out[t] = s3;
    }
}

// =======================================================================

extern "C" void kernel_launch(void* const* d_in, const int* in_sizes, int n_in,
                              void* d_out, int out_size, void* d_ws, size_t ws_size,
                              hipStream_t stream) {
    (void)in_sizes; (void)n_in; (void)out_size; (void)ws_size;
    const float* pc    = (const float*)d_in[0];
    const float* t_w1  = (const float*)d_in[1];
    const float* t_b1  = (const float*)d_in[2];
    const float* t_w2  = (const float*)d_in[3];
    const float* t_b2  = (const float*)d_in[4];
    const float* t_w3  = (const float*)d_in[5];
    const float* t_b3  = (const float*)d_in[6];
    const float* t_w4  = (const float*)d_in[7];
    const float* t_b4  = (const float*)d_in[8];
    const float* gc1_w = (const float*)d_in[9];
    const float* gc1_b = (const float*)d_in[10];
    const float* gc_w  = (const float*)d_in[11];
    const float* gc_b  = (const float*)d_in[12];
    const float* gp1_w = (const float*)d_in[13];
    const float* gp1_b = (const float*)d_in[14];
    const float* gp2_w = (const float*)d_in[15];
    const float* gp2_b = (const float*)d_in[16];
    const float* fc0_w = (const float*)d_in[17];
    const float* fc0_b = (const float*)d_in[18];
    const float* gatws = (const float*)d_in[19];
    const float* gatwd = (const float*)d_in[20];
    const float* gat_a = (const float*)d_in[21];
    const float* gat_b = (const float*)d_in[22];
    const float* fc1_w = (const float*)d_in[23];
    const float* fc1_b = (const float*)d_in[24];
    const float* fc2_w = (const float*)d_in[25];
    const float* fc2_b = (const float*)d_in[26];
    const float* fc3_w = (const float*)d_in[27];
    const float* fc3_b = (const float*)d_in[28];
    float* out = (float*)d_out;

    const int N = NPTS;
    char* p = (char*)d_ws;
    auto alloc = [&](size_t bytes) -> void* {
        void* r = (void*)p;
        p += (bytes + 255) & ~(size_t)255;
        return r;
    };
    float* qbuf  = (float*)alloc(9 * 4);
    float* x     = (float*)alloc((size_t)N * 3 * 4);
    float* pxy   = (float*)alloc((size_t)N * 2 * 4);
    float* pyz   = (float*)alloc((size_t)N * 2 * 4);
    float* pxz   = (float*)alloc((size_t)N * 2 * 4);
    int*   g0    = (int*)alloc((size_t)N * KMAIN * 4);
    int*   gxy   = (int*)alloc((size_t)N * KMAIN * 4);
    int*   gyz   = (int*)alloc((size_t)N * KMAIN * 4);
    int*   gxz   = (int*)alloc((size_t)N * KMAIN * 4);
    int*   gf0   = (int*)alloc((size_t)N * KMAIN * 4);
    int*   gf1   = (int*)alloc((size_t)N * KMAIN * 4);
    int*   gpl[3];
    for (int i = 0; i < 3; ++i) gpl[i] = (int*)alloc((size_t)N * KPLANE * 4);
    float* deg   = (float*)alloc((size_t)N * 4);
    float* rdg   = (float*)alloc((size_t)N * 4);
    float* Abuf  = (float*)alloc((size_t)N * 64 * 4);
    float* Bbuf  = (float*)alloc((size_t)N * 64 * 4);
    float* hp[3];
    for (int i = 0; i < 3; ++i) hp[i] = (float*)alloc((size_t)N * 16 * 4);
    float* H     = (float*)alloc((size_t)N * 1408 * 4);
    unsigned short* Hb   = (unsigned short*)alloc((size_t)N * 1408 * 2);
    float* t1f   = (float*)alloc((size_t)N * 64 * 4);
    unsigned short* t1b  = (unsigned short*)alloc((size_t)N * 64 * 2);
    unsigned short* w2tb = (unsigned short*)alloc((size_t)1024 * 64 * 2);
    unsigned short* fc0wb= (unsigned short*)alloc((size_t)1024 * 192 * 2);
    unsigned short* wsb  = (unsigned short*)alloc((size_t)1024 * 1408 * 2);
    unsigned short* wdb  = (unsigned short*)alloc((size_t)1024 * 1408 * 2);
    float* t2f   = (float*)alloc((size_t)N * 1024 * 4);   // reused: tnet fc2 out, fc0 out, gat out
    float* fsb   = (float*)alloc((size_t)N * 1024 * 4);
    float* fdb   = (float*)alloc((size_t)N * 1024 * 4);
    float* maxv  = (float*)alloc(1024 * 4);
    float* hvec  = (float*)alloc(1024 * 4);
    float* hg    = (float*)alloc(1024 * 4);

    const float rs20 = 1.0f / sqrtf((float)KMAIN);
    const float rs50 = 1.0f / sqrtf((float)KPLANE);
    dim3 gemmGrid(1024 / 64, N / 32);   // 32x64 output tile per wave

    auto run_rdeg = [&](const int* nbr, int k) {
        fzero_kernel<<<(N + 255) / 256, 256, 0, stream>>>(deg, N);
        degcount_kernel<<<(N * k + 255) / 256, 256, 0, stream>>>(nbr, N * k, deg);
        rdeg_kernel<<<(N + 255) / 256, 256, 0, stream>>>(deg, rdg, N);
    };

    // ---- T-net ----
    tnet_fc1_kernel<<<(N * 64 + 255) / 256, 256, 0, stream>>>(pc, t_w1, t_b1, t1f);
    f32_to_bf16_kernel<<<(N * 64 + 255) / 256, 256, 0, stream>>>(t1f, t1b, N * 64);
    transpose_to_bf16_kernel<<<(64 * 1024 + 255) / 256, 256, 0, stream>>>(t_w2, w2tb, 64, 1024);
    wmma_gemm_bf16<<<gemmGrid, 32, 0, stream>>>(t1b, 64, w2tb, 64, t_b2, t2f, 1024, 64);
    colmax_kernel<<<1024, 256, 0, stream>>>(t2f, N, 1024, maxv);
    tnet_head_qr_kernel<<<1, 256, 0, stream>>>(maxv, t_w3, t_b3, t_w4, t_b4, qbuf, out + 40);
    rotate_project_kernel<<<(N + 255) / 256, 256, 0, stream>>>(pc, qbuf, x, pxy, pyz, pxz);

    // ---- kNN graphs on xyz + projections ----
    knn_kernel<<<N, 256, 0, stream>>>(x,   3, 3, KMAIN, g0);
    knn_kernel<<<N, 256, 0, stream>>>(pxy, 2, 2, KMAIN, gxy);
    knn_kernel<<<N, 256, 0, stream>>>(pyz, 2, 2, KMAIN, gyz);
    knn_kernel<<<N, 256, 0, stream>>>(pxz, 2, 2, KMAIN, gxz);

    // ---- main branch ----
    run_rdeg(g0, KMAIN);
    gconv_kernel<<<N, 64, 0, stream>>>(g0, KMAIN, x, 3, rdg, gc1_w, gc1_b, 3, 64, Abuf, 64, rs20);
    gconv_kernel<<<N, 64, 0, stream>>>(g0, KMAIN, Abuf, 64, rdg, gc_w + 0 * 4096, gc_b + 0 * 64,
                                       64, 64, H + 0, 1408, rs20);
    knn_kernel<<<N, 256, 0, stream>>>(H + 0, 1408, 64, KMAIN, gf0);
    run_rdeg(gf0, KMAIN);
    gconv_kernel<<<N, 64, 0, stream>>>(gf0, KMAIN, H + 0, 1408, rdg, gc_w + 1 * 4096, gc_b + 1 * 64,
                                       64, 64, Bbuf, 64, rs20);
    gconv_kernel<<<N, 64, 0, stream>>>(gf0, KMAIN, Bbuf, 64, rdg, gc_w + 2 * 4096, gc_b + 2 * 64,
                                       64, 64, H + 64, 1408, rs20);
    knn_kernel<<<N, 256, 0, stream>>>(H + 64, 1408, 64, KMAIN, gf1);
    run_rdeg(gf1, KMAIN);
    gconv_kernel<<<N, 64, 0, stream>>>(gf1, KMAIN, H + 64, 1408, rdg, gc_w + 3 * 4096, gc_b + 3 * 64,
                                       64, 64, H + 128, 1408, rs20);

    // ---- fc0 + global max + broadcast (WMMA on first 192 cols of H) ----
    f32_to_bf16_kernel<<<(N * 1408 + 255) / 256, 256, 0, stream>>>(H, Hb, N * 1408);
    transpose_to_bf16_kernel<<<(192 * 1024 + 255) / 256, 256, 0, stream>>>(fc0_w, fc0wb, 192, 1024);
    wmma_gemm_bf16<<<gemmGrid, 32, 0, stream>>>(Hb, 1408, fc0wb, 192, fc0_b, t2f, 1024, 192);
    colmax_kernel<<<1024, 256, 0, stream>>>(t2f, N, 1024, hvec);
    broadcast_cols_kernel<<<(N * 1024 + 255) / 256, 256, 0, stream>>>(hvec, H, 1408, 192, 1024);

    // ---- plane branches ----
    const int* pg[3] = {gxy, gyz, gxz};
    for (int i = 0; i < 3; ++i) {
        run_rdeg(pg[i], KMAIN);
        gconv_kernel<<<N, 64, 0, stream>>>(pg[i], KMAIN, x, 3, rdg,
                                           gp1_w + i * 48, gp1_b + i * 16, 3, 16, hp[i], 16, rs20);
        knn_kernel<<<N, 256, 0, stream>>>(hp[i], 16, 16, KPLANE, gpl[i]);
        run_rdeg(gpl[i], KPLANE);
        gconv_kernel<<<N, 64, 0, stream>>>(gpl[i], KPLANE, hp[i], 16, rdg,
                                           gp2_w + i * 1024, gp2_b + i * 64, 16, 64,
                                           H + 1216 + 64 * i, 1408, rs50);
    }

    // ---- GATv2 projections (WMMA, K=1408) ----
    f32_to_bf16_kernel<<<(N * 1408 + 255) / 256, 256, 0, stream>>>(H, Hb, N * 1408);
    transpose_to_bf16_kernel<<<(1408 * 1024 + 255) / 256, 256, 0, stream>>>(gatws, wsb, 1408, 1024);
    transpose_to_bf16_kernel<<<(1408 * 1024 + 255) / 256, 256, 0, stream>>>(gatwd, wdb, 1408, 1024);
    wmma_gemm_bf16<<<gemmGrid, 32, 0, stream>>>(Hb, 1408, wsb, 1408, nullptr, fsb, 1024, 1408);
    wmma_gemm_bf16<<<gemmGrid, 32, 0, stream>>>(Hb, 1408, wdb, 1408, nullptr, fdb, 1024, 1408);

    // ---- attention + global max + head ----
    gatv2_kernel<<<N, 256, 0, stream>>>(g0, fsb, fdb, gat_a, gat_b, t2f);
    colmax_kernel<<<1024, 256, 0, stream>>>(t2f, N, 1024, hg);
    head_kernel<<<1, 256, 0, stream>>>(hg, fc1_w, fc1_b, fc2_w, fc2_b, fc3_w, fc3_b, out);
}
